// TGF_M_72275709657191
// MI455X (gfx1250) — matrix operand order repaired and checked by reading, not compiled, not run on hardware
//
#include <hip/hip_runtime.h>

#define D_ 256
#define HD_ 16
#define G_ 16
#define HK_ 4
#define N_ 16384
#define E_ 49152
#define B_ 256
#define VA_ 128
#define VB_ 8
#define SQRT2PI_ 2.5066270f

typedef __attribute__((ext_vector_type(16))) _Float16 v16h;
typedef __attribute__((ext_vector_type(8)))  float    v8f;

// ---------------------------------------------------------------- utility
__global__ void k_zero(float* __restrict__ p, int n) {
  int i = blockIdx.x * 256 + threadIdx.x;
  if (i < n) p[i] = 0.f;
}
__global__ void k_zero_i(int* __restrict__ p, int n) {
  int i = blockIdx.x * 256 + threadIdx.x;
  if (i < n) p[i] = 0;
}
__global__ void k_add(const float* __restrict__ a, const float* __restrict__ b,
                      float* __restrict__ o, int n) {
  int i = blockIdx.x * 256 + threadIdx.x;
  if (i < n) o[i] = a[i] + b[i];
}
// f32 -> f16 pack, 4 elements/thread (sizes are multiples of 4)
__global__ void k_cvt(const float* __restrict__ f, _Float16* __restrict__ h, int n) {
  int i = (blockIdx.x * 256 + threadIdx.x) * 4;
  if (i < n) {
    float4 v = *(const float4*)(f + i);
    h[i]     = (_Float16)v.x;
    h[i + 1] = (_Float16)v.y;
    h[i + 2] = (_Float16)v.z;
    h[i + 3] = (_Float16)v.w;
  }
}

// ---------------------------------------------------------------- degree
__global__ void k_deg_count(const int* __restrict__ dst, int* __restrict__ cnt) {
  int e = blockIdx.x * 256 + threadIdx.x;
  if (e < E_) atomicAdd(&cnt[dst[e]], 1);
}
__global__ void k_deg_fin(int* __restrict__ cnt) {
  int n = blockIdx.x * 256 + threadIdx.x;
  if (n < N_) {
    int d = cnt[n] - 1;
    cnt[n] = d < 0 ? 0 : (d > 3 ? 3 : d);
  }
}

// ------------------------------------------------- edge feature fuse (both)
__global__ void k_edge_gate(const int* __restrict__ eidx, const int* __restrict__ eattr,
                            const float* __restrict__ dist, const float* __restrict__ bond2d,
                            const float* __restrict__ gmean, const float* __restrict__ gstd,
                            const float* __restrict__ attn_w, const float* __restrict__ attn_b,
                            float* __restrict__ acc) {
  int e = blockIdx.x;
  int c = threadIdx.x;
  int dst = eidx[E_ + e];
  float h2d = 0.f;
  #pragma unroll
  for (int f = 0; f < 3; ++f) {
    int idx = eattr[e * 3 + f];
    h2d += bond2d[(f * VB_ + idx) * D_ + c];
  }
  float sd = fabsf(gstd[c]) + 0.01f;
  float df = (dist[e] - gmean[c]) / sd;
  float h3d = expf(-0.5f * df * df) / (SQRT2PI_ * sd);
  __shared__ float red[256];
  red[c] = attn_w[c] * h2d + attn_w[D_ + c] * h3d;
  __syncthreads();
  for (int s = 128; s > 0; s >>= 1) {
    if (c < s) red[c] += red[c + s];
    __syncthreads();
  }
  float w = 1.f / (1.f + expf(-(red[0] + attn_b[0])));
  atomicAdd(&acc[(size_t)dst * D_ + c], w * h2d + (1.f - w) * h3d);
}

__global__ void k_node_init(const int* __restrict__ xf, const float* __restrict__ aemb,
                            const int* __restrict__ deg, const float* __restrict__ scale_both,
                            const float* __restrict__ acc, float* __restrict__ hin) {
  int i = blockIdx.x * 256 + threadIdx.x;
  if (i >= N_ * D_) return;
  int n = i >> 8, c = i & 255;
  float s = 0.f;
  #pragma unroll
  for (int f = 0; f < 9; ++f) {
    int idx = xf[n * 9 + f];
    s += aemb[(f * VA_ + idx) * D_ + c];
  }
  hin[i] = s + expf(scale_both[deg[n] * D_ + c]) * acc[i];
}

// ---------------------------------------------------------------- WMMA GEMM
// out[R,O] = A[R,K] @ W[O,K]^T via v_wmma_f32_16x16x32_f16 on pre-packed f16.
// One wave computes a 16x64 strip: 4 accumulators reuse one A fragment.
// R mult of 16, K mult of 32, O mult of 64.
// MODE 0: out = acc + bias
// MODE 1: atomicAdd(out[rowmap[r]*O+c], acc + bias)   (fused segment_sum)
// MODE 2: out += expf(lscale[c]) * (acc + bias[c])
template <int MODE>
__global__ void k_gemm16(const _Float16* __restrict__ A, const _Float16* __restrict__ W,
                         float* __restrict__ out, const float* __restrict__ bias,
                         const float* __restrict__ lscale, const int* __restrict__ rowmap,
                         int R, int K, int O) {
  int gw = (blockIdx.x * 256 + threadIdx.x) >> 5;
  int lane = threadIdx.x & 31;
  int nstrips = O >> 6;
  int mt = gw / nstrips;
  int ns = gw - mt * nstrips;
  if (mt * 16 >= R) return;

  int rA = mt * 16 + (lane & 15);
  int cB = ns * 64 + (lane & 15);   // W row feeding n-tile 0 of the strip
  int ka0 = (lane < 16) ? 0 : 8;    // A layout: lanes 0-15: K 0-7/16-23; 16-31: 8-15/24-31
  int kb0 = (lane < 16) ? 0 : 8;    // B layout: lanes 0-15: K 0-7/16-23; 16-31: 8-15/24-31

  v8f acc[4];
  #pragma unroll
  for (int t = 0; t < 4; ++t) acc[t] = (v8f){};

  for (int kk = 0; kk < K; kk += 32) {
    const _Float16* ap = A + (size_t)rA * K + kk + ka0;
    v16h a;
    #pragma unroll
    for (int e = 0; e < 8; ++e) {
      a[e]     = ap[e];
      a[e + 8] = ap[16 + e];
    }
    const _Float16* bp = W + (size_t)cB * K + kk + kb0;
    #pragma unroll
    for (int t = 0; t < 4; ++t) {
      const _Float16* bt = bp + (size_t)t * 16 * K;
      v16h b;
      #pragma unroll
      for (int e = 0; e < 8; ++e) { b[e] = bt[e]; b[e + 8] = bt[16 + e]; }
      acc[t] = __builtin_amdgcn_wmma_f32_16x16x32_f16(false, a, false, b, (short)0,
                                                      acc[t], false, false);
    }
  }

  int m0 = mt * 16 + ((lane >> 4) << 3);  // C/D: M = r + 8*(lane/16), N = lane%16
  int rows[8];
  if (MODE == 1) {
    #pragma unroll
    for (int r = 0; r < 8; ++r) rows[r] = rowmap[m0 + r];
  }
  #pragma unroll
  for (int t = 0; t < 4; ++t) {
    int col = ns * 64 + t * 16 + (lane & 15);
    float bv = bias ? bias[col] : 0.f;
    float sc = (MODE == 2) ? expf(lscale[col]) : 0.f;
    #pragma unroll
    for (int r = 0; r < 8; ++r) {
      float v = acc[t][r] + bv;
      if (MODE == 0) {
        out[(size_t)(m0 + r) * O + col] = v;
      } else if (MODE == 1) {
        atomicAdd(&out[(size_t)rows[r] * O + col], v);
      } else {
        out[(size_t)(m0 + r) * O + col] += sc * v;
      }
    }
  }
}

// ----------------------------------------- conv glb2: GN + grouped gate/val
// grid (E/256, G): blockIdx.y = group; per-group 16x16 weights staged in LDS
__global__ void k_conv_edge(const float* __restrict__ xp, const int* __restrict__ eidx,
                            const int* __restrict__ eattr, const float* __restrict__ cbond,
                            const float* __restrict__ gw, const float* __restrict__ vw,
                            float* __restrict__ t) {
  __shared__ float sgw[256], svw[256];
  int g = blockIdx.y;
  sgw[threadIdx.x] = gw[g * 256 + threadIdx.x];
  svw[threadIdx.x] = vw[g * 256 + threadIdx.x];
  __syncthreads();
  int e = blockIdx.x * 256 + threadIdx.x;
  int src = eidx[e], dst = eidx[E_ + e];

  float bond[16];
  #pragma unroll
  for (int d = 0; d < 16; ++d) bond[d] = 0.f;
  #pragma unroll
  for (int f = 0; f < 3; ++f) {
    const float* bp = cbond + ((size_t)(f * VB_ + eattr[e * 3 + f])) * D_ + g * 16;
    #pragma unroll
    for (int d = 0; d < 16; ++d) bond[d] += bp[d];
  }
  float gi[16], vi[16];
  const float* xg = xp + (size_t)dst * D_ + g * 16;
  const float* xs = xp + (size_t)src * D_ + g * 16;
  float mg = 0.f, mv = 0.f;
  #pragma unroll
  for (int d = 0; d < 16; ++d) {
    gi[d] = xg[d] + bond[d]; mg += gi[d];
    vi[d] = xs[d] + bond[d]; mv += vi[d];
  }
  mg *= (1.f / 16.f); mv *= (1.f / 16.f);
  float vg = 0.f, vv = 0.f;
  #pragma unroll
  for (int d = 0; d < 16; ++d) {
    float a = gi[d] - mg; vg += a * a;
    float b = vi[d] - mv; vv += b * b;
  }
  float rg = rsqrtf(vg * (1.f / 16.f) + 1e-5f);
  float rv = rsqrtf(vv * (1.f / 16.f) + 1e-5f);
  #pragma unroll
  for (int d = 0; d < 16; ++d) {
    gi[d] = (gi[d] - mg) * rg;
    vi[d] = (vi[d] - mv) * rv;
  }
  float* to = t + (size_t)e * D_ + g * 16;
  #pragma unroll
  for (int o = 0; o < 16; ++o) {
    float sg = 0.f, sv = 0.f;
    #pragma unroll
    for (int d = 0; d < 16; ++d) {
      sg += gi[d] * sgw[o * 16 + d];
      sv += vi[d] * svw[o * 16 + d];
    }
    to[o] = fmaxf(sg, 0.f) * sv;
  }
}

// ----------------- glb middle: GN + grouped gate/val (DOUT outputs / group)
// grid (R/256, G); per-group weights staged in LDS
__global__ void k_glb_mid(const float* __restrict__ x, const float* __restrict__ gw,
                          const float* __restrict__ vw, float* __restrict__ t,
                          int DOUT) {
  __shared__ float sgw[48 * 16], svw[48 * 16];
  int g = blockIdx.y;
  for (int i = threadIdx.x; i < DOUT * 16; i += 256) {
    sgw[i] = gw[(size_t)g * DOUT * 16 + i];
    svw[i] = vw[(size_t)g * DOUT * 16 + i];
  }
  __syncthreads();
  int r = blockIdx.x * 256 + threadIdx.x;
  float xx[16];
  const float* xpp = x + (size_t)r * D_ + g * 16;
  float m = 0.f;
  #pragma unroll
  for (int d = 0; d < 16; ++d) { xx[d] = xpp[d]; m += xx[d]; }
  m *= (1.f / 16.f);
  float v = 0.f;
  #pragma unroll
  for (int d = 0; d < 16; ++d) { float a = xx[d] - m; v += a * a; }
  float rs = rsqrtf(v * (1.f / 16.f) + 1e-5f);
  #pragma unroll
  for (int d = 0; d < 16; ++d) xx[d] = (xx[d] - m) * rs;
  float* to = t + (size_t)r * (G_ * DOUT) + g * DOUT;
  for (int o = 0; o < DOUT; ++o) {
    float sg = 0.f, sv = 0.f;
    #pragma unroll
    for (int d = 0; d < 16; ++d) {
      sg += xx[d] * sgw[o * 16 + d];
      sv += xx[d] * svw[o * 16 + d];
    }
    to[o] = fmaxf(sg, 0.f) * sv;
  }
}

// -------------------------------- attention q/k/v (GN fused, exp transforms)
// grid (N/256, G); per-group q/k/v weights staged in LDS
__global__ void k_att_qkv(const float* __restrict__ x, const float* __restrict__ qw,
                          const float* __restrict__ kw, const float* __restrict__ vw,
                          float* __restrict__ Q, float* __restrict__ Kb,
                          float* __restrict__ Vb) {
  __shared__ float sq[512], sk[512], sv[512];
  int g = blockIdx.y;
  for (int i = threadIdx.x; i < 512; i += 256) {
    sq[i] = qw[(size_t)g * 512 + i];
    sk[i] = kw[(size_t)g * 512 + i];
    sv[i] = vw[(size_t)g * 512 + i];
  }
  __syncthreads();
  int n = blockIdx.x * 256 + threadIdx.x;
  float xx[16];
  const float* xpp = x + (size_t)n * D_ + g * 16;
  float m = 0.f;
  #pragma unroll
  for (int d = 0; d < 16; ++d) { xx[d] = xpp[d]; m += xx[d]; }
  m *= (1.f / 16.f);
  float v = 0.f;
  #pragma unroll
  for (int d = 0; d < 16; ++d) { float a = xx[d] - m; v += a * a; }
  float rs = rsqrtf(v * (1.f / 16.f) + 1e-5f);
  #pragma unroll
  for (int d = 0; d < 16; ++d) xx[d] = (xx[d] - m) * rs;
  size_t base = (size_t)n * 512 + g * 32;
  for (int o = 0; o < 32; ++o) {
    float aq = 0.f, ak = 0.f, av = 0.f;
    #pragma unroll
    for (int d = 0; d < 16; ++d) {
      aq += xx[d] * sq[o * 16 + d];
      ak += xx[d] * sk[o * 16 + d];
      av += xx[d] * sv[o * 16 + d];
    }
    Q[base + o]  = expf(aq * 0.25f);   // / sqrt(HD=16)
    Kb[base + o] = expf(ak * 0.25f);
    Vb[base + o] = av;
  }
}

__global__ void k_seg_nodes_add(const float* __restrict__ hin, const int* __restrict__ batch,
                                float* __restrict__ hv) {
  int i = blockIdx.x * 256 + threadIdx.x;
  if (i >= N_ * D_) return;
  int n = i >> 8, c = i & 255;
  atomicAdd(&hv[(size_t)batch[n] * D_ + c], hin[i]);
}
__global__ void k_seg_sum_512(const float* __restrict__ Kb, const int* __restrict__ batch,
                              float* __restrict__ ks) {
  int i = blockIdx.x * 256 + threadIdx.x;
  if (i >= N_ * 512) return;
  int n = i >> 9, j = i & 511;
  atomicAdd(&ks[(size_t)batch[n] * 512 + j], Kb[i]);
}
__global__ void k_att_accum(const float* __restrict__ Kb, const float* __restrict__ Vb,
                            const int* __restrict__ batch, float* __restrict__ hatt) {
  int tid = blockIdx.x * 256 + threadIdx.x;
  if (tid >= N_ * 32) return;
  int n = tid >> 5, head = tid & 31;
  int b = batch[n];
  float kk[16], vv[16];
  size_t base = (size_t)n * 512 + head * 16;
  #pragma unroll
  for (int d = 0; d < 16; ++d) { kk[d] = Kb[base + d]; vv[d] = Vb[base + d]; }
  float* hp = hatt + ((size_t)b * 32 + head) * 256;
  #pragma unroll
  for (int h = 0; h < 16; ++h)
    #pragma unroll
    for (int w = 0; w < 16; ++w) atomicAdd(&hp[h * 16 + w], kk[h] * vv[w]);
}
__global__ void k_att_out(const float* __restrict__ Q, const float* __restrict__ ks,
                          const float* __restrict__ hatt, const int* __restrict__ batch,
                          float* __restrict__ AO) {
  int tid = blockIdx.x * 256 + threadIdx.x;
  if (tid >= N_ * 32) return;
  int n = tid >> 5, head = tid & 31;
  int b = batch[n];
  float q[16];
  size_t base = (size_t)n * 512 + head * 16;
  const float* kp = ks + (size_t)b * 512 + head * 16;
  float den = 0.f;
  #pragma unroll
  for (int h = 0; h < 16; ++h) { q[h] = Q[base + h]; den += q[h] * kp[h]; }
  float inv = 1.f / den;
  const float* hp = hatt + ((size_t)b * 32 + head) * 256;
  #pragma unroll
  for (int w = 0; w < 16; ++w) {
    float s = 0.f;
    #pragma unroll
    for (int h = 0; h < 16; ++h) s += q[h] * hp[h * 16 + w];
    AO[base + w] = inv * s;
  }
}

__global__ void k_axpy_deg(const int* __restrict__ deg, const float* __restrict__ cscale,
                           const float* __restrict__ xraw, float* __restrict__ xout) {
  int i = blockIdx.x * 256 + threadIdx.x;
  if (i >= N_ * D_) return;
  int n = i >> 8, c = i & 255;
  xout[i] += expf(cscale[deg[n] * D_ + c]) * xraw[i];
}
__global__ void k_axpy_batch(const int* __restrict__ batch, const float* __restrict__ vscale,
                             const float* __restrict__ vout, float* __restrict__ hout) {
  int i = blockIdx.x * 256 + threadIdx.x;
  if (i >= N_ * D_) return;
  int n = i >> 8, c = i & 255;
  hout[i] += expf(vscale[c]) * vout[(size_t)batch[n] * D_ + c];
}

// ================================================================= host
static inline int nblk(long long t) { return (int)((t + 255) / 256); }

static void gemm(const float* A, const float* W, float* out, const float* bias,
                 const float* lscale, const int* rowmap, int R, int K, int O, int mode,
                 _Float16* Ah, _Float16* Wh, hipStream_t s) {
  // pack operands to f16 once (removes redundant per-wave cvt + halves loop bytes)
  k_cvt<<<nblk(((long long)R * K) / 4), 256, 0, s>>>(A, Ah, R * K);
  k_cvt<<<nblk(((long long)O * K) / 4), 256, 0, s>>>(W, Wh, O * K);
  long long thr = (long long)(R / 16) * (O >> 6) * 32;
  int g = nblk(thr);
  if (mode == 0)
    k_gemm16<0><<<g, 256, 0, s>>>(Ah, Wh, out, bias, lscale, rowmap, R, K, O);
  else if (mode == 1)
    k_gemm16<1><<<g, 256, 0, s>>>(Ah, Wh, out, bias, lscale, rowmap, R, K, O);
  else
    k_gemm16<2><<<g, 256, 0, s>>>(Ah, Wh, out, bias, lscale, rowmap, R, K, O);
}

extern "C" void kernel_launch(void* const* d_in, const int* in_sizes, int n_in,
                              void* d_out, int out_size, void* d_ws, size_t ws_size,
                              hipStream_t stream) {
  (void)in_sizes; (void)n_in; (void)out_size; (void)ws_size;
  const int*   x_feat   = (const int*)d_in[0];
  const int*   eidx     = (const int*)d_in[1];
  const int*   eattr    = (const int*)d_in[2];
  const int*   batch    = (const int*)d_in[3];
  const float* dist     = (const float*)d_in[5];
  const float* atom_emb = (const float*)d_in[6];
  const float* bond2d   = (const float*)d_in[7];
  const float* gmean    = (const float*)d_in[8];
  const float* gstd     = (const float*)d_in[9];
  const float* attn_w   = (const float*)d_in[10];
  const float* attn_b   = (const float*)d_in[11];
  const float* sboth    = (const float*)d_in[12];
  const float* cbond    = (const float*)d_in[13];
  const float* cprew    = (const float*)d_in[14];
  const float* cgatew   = (const float*)d_in[15];
  const float* cvalw    = (const float*)d_in[16];
  const float* cpostw   = (const float*)d_in[17];
  const float* cpostb   = (const float*)d_in[18];
  const float* cscale   = (const float*)d_in[19];
  const float* vprew    = (const float*)d_in[20];
  const float* vpreb    = (const float*)d_in[21];
  const float* vgatew   = (const float*)d_in[22];
  const float* vvalw    = (const float*)d_in[23];
  const float* vpostw   = (const float*)d_in[24];
  const float* vpostb   = (const float*)d_in[25];
  const float* vscale   = (const float*)d_in[26];
  const float* aprew    = (const float*)d_in[27];
  const float* apreb    = (const float*)d_in[28];
  const float* aqw      = (const float*)d_in[29];
  const float* akw      = (const float*)d_in[30];
  const float* avw      = (const float*)d_in[31];
  const float* apostw   = (const float*)d_in[32];
  const float* apostb   = (const float*)d_in[33];
  const float* ascale   = (const float*)d_in[34];
  const float* mprew    = (const float*)d_in[35];
  const float* mpreb    = (const float*)d_in[36];
  const float* mgatew   = (const float*)d_in[37];
  const float* mvalw    = (const float*)d_in[38];
  const float* mpostw   = (const float*)d_in[39];
  const float* mpostb   = (const float*)d_in[40];
  const int*   dst      = eidx + E_;

  // ---- workspace layout (floats) ----
  float* F = (float*)d_ws;
  size_t o = 0;
  float* NB0 = F + o; o += (size_t)N_ * D_;        // h_in
  float* NB1 = F + o; o += (size_t)N_ * D_;        // x_raw / h_out
  float* NB2 = F + o; o += (size_t)N_ * D_;        // acc / x_out
  float* NB3 = F + o; o += (size_t)N_ * D_;        // pre-activation scratch
  float* U   = F + o; o += (size_t)3 * N_ * 512;   // edge-t / Q,K,V / main-t
  float* hv  = F + o; o += (size_t)B_ * D_;        // virtual node state
  float* hatt= F + o; o += (size_t)B_ * 32 * 256;  // attention kv state
  float* ks  = F + o; o += (size_t)B_ * 512;       // k segment sums
  float* vpre= F + o; o += (size_t)B_ * D_;
  float* vt  = F + o; o += (size_t)B_ * 512;
  float* vou = F + o; o += (size_t)B_ * D_;
  _Float16* Ah = (_Float16*)(F + o); o += (size_t)N_ * 768 / 2;  // f16 A staging (max R*K)
  _Float16* Wh = (_Float16*)(F + o); o += (size_t)D_ * 768 / 2;  // f16 W staging (max O*K)
  int*   deg = (int*)(F + o);
  float* Q   = U;
  float* Kb  = U + (size_t)N_ * 512;
  float* Vb  = U + (size_t)2 * N_ * 512;
  float* AO  = Kb;                                  // reuse K after it is consumed

  const int ND = N_ * D_;

  // ---- init / zero persistent accumulators (harness does not re-poison) ----
  k_zero<<<nblk(ND), 256, 0, stream>>>(NB2, ND);
  k_zero<<<nblk((long long)B_ * D_), 256, 0, stream>>>(hv, B_ * D_);
  k_zero<<<nblk((long long)B_ * 32 * 256), 256, 0, stream>>>(hatt, B_ * 32 * 256);
  k_zero_i<<<nblk(N_), 256, 0, stream>>>(deg, N_);
  k_deg_count<<<nblk(E_), 256, 0, stream>>>(dst, deg);
  k_deg_fin<<<nblk(N_), 256, 0, stream>>>(deg);
  k_edge_gate<<<E_, 256, 0, stream>>>(eidx, eattr, dist, bond2d, gmean, gstd,
                                      attn_w, attn_b, NB2);
  k_node_init<<<nblk(ND), 256, 0, stream>>>(x_feat, atom_emb, deg, sboth, NB2, NB0);

  for (int l = 0; l < 2; ++l) {
    // ---------------- ConvMessage ----------------
    k_zero<<<nblk(ND), 256, 0, stream>>>(NB2, ND);   // x_out = 0
    const float* xraw = NB0;                         // x_raw = h_in
    for (int k = 0; k < HK_; ++k) {
      int lk = l * HK_ + k;
      if (k > 0 && (k % 2) == 0) {                   // hop reset
        k_add<<<nblk(ND), 256, 0, stream>>>(NB0, NB2, NB1, ND);
        k_zero<<<nblk(ND), 256, 0, stream>>>(NB2, ND);
        xraw = NB1;
      }
      gemm(xraw, cprew + (size_t)lk * D_ * D_, NB3, nullptr, nullptr, nullptr,
           N_, D_, D_, 0, Ah, Wh, stream);           // xp = x_raw @ pre^T
      k_zero<<<nblk(ND), 256, 0, stream>>>(NB1, ND); // segment-sum target
      k_conv_edge<<<dim3(E_ / 256, G_), 256, 0, stream>>>(
          NB3, eidx, eattr, cbond + (size_t)lk * 3 * VB_ * D_,
          cgatew + (size_t)lk * G_ * 256, cvalw + (size_t)lk * G_ * 256, U);
      gemm(U, cpostw + (size_t)lk * D_ * D_, NB1, cpostb + (size_t)lk * D_,
           nullptr, dst, E_, D_, D_, 1, Ah, Wh, stream);  // msg + fused scatter
      k_axpy_deg<<<nblk(ND), 256, 0, stream>>>(deg, cscale + (size_t)lk * 4 * D_,
                                               NB1, NB2);
      xraw = NB1;
    }
    k_add<<<nblk(ND), 256, 0, stream>>>(NB0, NB2, NB1, ND);  // h_out

    // ---------------- VirtMessage ----------------
    k_seg_nodes_add<<<nblk(ND), 256, 0, stream>>>(NB0, batch, hv);
    gemm(hv, vprew + (size_t)l * D_ * D_, vpre, vpreb + (size_t)l * D_,
         nullptr, nullptr, B_, D_, D_, 0, Ah, Wh, stream);
    k_glb_mid<<<dim3(B_ / 256, G_), 256, 0, stream>>>(
        vpre, vgatew + (size_t)l * G_ * 32 * HD_, vvalw + (size_t)l * G_ * 32 * HD_,
        vt, 32);
    gemm(vt, vpostw + (size_t)l * D_ * 512, vou, vpostb + (size_t)l * D_,
         nullptr, nullptr, B_, 512, D_, 0, Ah, Wh, stream);
    k_axpy_batch<<<nblk(ND), 256, 0, stream>>>(batch, vscale + (size_t)l * D_, vou, NB1);

    // ---------------- AttMessage ----------------
    gemm(NB0, aprew + (size_t)l * D_ * D_, NB3, apreb + (size_t)l * D_,
         nullptr, nullptr, N_, D_, D_, 0, Ah, Wh, stream);
    k_att_qkv<<<dim3(N_ / 256, G_), 256, 0, stream>>>(
        NB3, aqw + (size_t)l * G_ * 32 * HD_, akw + (size_t)l * G_ * 32 * HD_,
        avw + (size_t)l * G_ * 32 * HD_, Q, Kb, Vb);
    k_zero<<<nblk((long long)B_ * 512), 256, 0, stream>>>(ks, B_ * 512);
    k_seg_sum_512<<<nblk((long long)N_ * 512), 256, 0, stream>>>(Kb, batch, ks);
    k_att_accum<<<nblk((long long)N_ * 32), 256, 0, stream>>>(Kb, Vb, batch, hatt);
    k_att_out<<<nblk((long long)N_ * 32), 256, 0, stream>>>(Q, ks, hatt, batch, AO);
    gemm(AO, apostw + (size_t)l * D_ * 512, NB1, apostb + (size_t)l * D_,
         ascale + (size_t)l * D_, nullptr, N_, 512, D_, 2, Ah, Wh, stream);

    // ---------------- main gated block ----------------
    gemm(NB1, mprew + (size_t)l * D_ * D_, NB3, mpreb + (size_t)l * D_,
         nullptr, nullptr, N_, D_, D_, 0, Ah, Wh, stream);
    k_glb_mid<<<dim3(N_ / 256, G_), 256, 0, stream>>>(
        NB3, mgatew + (size_t)l * G_ * 48 * HD_, mvalw + (size_t)l * G_ * 48 * HD_,
        U, 48);
    gemm(U, mpostw + (size_t)l * D_ * 768,
         (l == 0) ? NB0 : (float*)d_out, mpostb + (size_t)l * D_,
         nullptr, nullptr, N_, 768, D_, 0, Ah, Wh, stream);
  }
}
